// CombSubFastFacV3_66821101191259
// MI455X (gfx1250) — matrix operand order repaired and checked by reading, not compiled, not run on hardware
//
#include <hip/hip_runtime.h>
#include <hip/hip_bf16.h>
#include <hip/hip_fp16.h>

// ---------------------------------------------------------------------------
// CombSub synth for MI455X (gfx1250, wave32).
//
// HBM-bound problem (~270MB traffic -> ~12us floor at 23.3TB/s). One pass:
// per (batch,frame) wave computes rfft(comb), rfft(noise) via a four-step
// 1024 = 32x32 FFT whose 32-point DFT stages are 32x32 matmuls mapped onto
// V_WMMA_F32_16X16X32_F16 (K=32 matches exactly), applies spectral filters,
// runs the inverse FFT the same way, overlap-adds with global_atomic_add_f32.
//
// This revision adds the Tensor Data Mover path: interior frames are DMA'd
// global->LDS with tensor_load_to_lds (1-D D#, tile_dim0=1024, f32), using
// TDM's out-of-bounds zero-fill for the right-edge frame, synchronized with
// s_wait_tensorcnt. t==0 (negative start) falls back to the manual load.
// ---------------------------------------------------------------------------

#define BATCH 8
#define NFR   2048            // f0 frames
#define BS    512
#define NSMP  (NFR * BS)      // 1048576 samples per batch
#define NWIN  1024            // 2*BS FFT length
#define NBINS 513
#define FSR   44100.0f
#define PI_F  3.14159265358979f

typedef __attribute__((ext_vector_type(16))) _Float16 v16h;
typedef __attribute__((ext_vector_type(8)))  float    v8f;
typedef __attribute__((ext_vector_type(4)))  unsigned int u32x4;
typedef __attribute__((ext_vector_type(4)))  int      i32x4;
typedef __attribute__((ext_vector_type(8)))  int      i32x8;

union H16 { v16h v; _Float16 e[16]; };

#define WMMA(a, b, c) __builtin_amdgcn_wmma_f32_16x16x32_f16( \
    false, (a), false, (b), (short)0, (c), false, false)

#if __has_builtin(__builtin_amdgcn_tensor_load_to_lds) && \
    __has_builtin(__builtin_amdgcn_s_wait_tensorcnt)
#define HAVE_TDM 1
#else
#define HAVE_TDM 0
#endif

#if HAVE_TDM
// 1-D TDM load: 1024 f32 from gsrc into LDS at ldst; elements past
// valid_elems are zero-filled by TDM OOB handling (ISA ch.8: reads beyond
// tensor dimension return zero). One descriptor per wave; TENSORcnt tracked.
__device__ inline void tdm_load_frame(const float* gsrc, float* ldst,
                                      int valid_elems) {
  const unsigned long long ga = (unsigned long long)(size_t)gsrc;
  const unsigned int lds = (unsigned int)(size_t)ldst;  // AS(3) offset bits
  // D# group 0: count=1 | lds_addr | global_addr[56:0] | type=2
  const u32x4 g0 = {1u, lds, (unsigned int)ga,
                    (unsigned int)((ga >> 32) & 0x01FFFFFFull) | (2u << 30)};
  // D# group 1: data_size=2 (4B), no multicast/pad/iterate;
  // tensor_dim0 = valid_elems (bits 79:48), tile_dim0 = 1024 (bits 127:112),
  // tile_dim1 = 0 (1-D), tensor_dim0_stride (bits 207:160) unused for 1-D.
  const i32x8 g1 = {(int)(2u << 16),
                    (int)((valid_elems & 0xFFFF) << 16),
                    (int)((valid_elems >> 16) & 0xFFFF),
                    (int)(1024u << 16),
                    0,
                    valid_elems,
                    0,
                    0};
  const i32x4 z4 = {0, 0, 0, 0};
#if defined(__clang_major__) && __clang_major__ >= 23
  const i32x8 z8 = {0, 0, 0, 0, 0, 0, 0, 0};
  __builtin_amdgcn_tensor_load_to_lds(g0, g1, z4, z4, z8, 0);
#else
  __builtin_amdgcn_tensor_load_to_lds(g0, g1, z4, z4, 0);
#endif
}
#endif

// Load one windowed 1024-sample frame into LDS. Interior frames use the TDM
// (wave-uniform branch; TDM ignores EXEC), t==0 uses the manual path; the right edge uses TDM zero-fill.
__device__ inline void load_frame(const float* __restrict__ gbase,
                                  float* __restrict__ dst, int t,
                                  const float* __restrict__ Win, int lane) {
#if HAVE_TDM
  if (t != 0) {
    const int start = t * BS - BS;  // >= 0
    tdm_load_frame(gbase + start, dst, NSMP - start);
    __builtin_amdgcn_s_wait_tensorcnt(0);
    for (int j = lane; j < NWIN; j += 32) dst[j] *= Win[j];
    return;
  }
#endif
  for (int j = lane; j < NWIN; j += 32) {
    const int s = t * BS + j - BS;
    const float v = (s >= 0 && s < NSMP) ? gbase[s] : 0.0f;
    dst[j] = v * Win[j];
  }
}

// ---- WMMA operand packing (layouts per CDNA5 ISA 7.12.2, wave32) ----------
// All lane-dependent indexing is branch-free bit arithmetic (keeps EXEC full
// around WMMA, which the ISA requires).

// A (16x32 f16): lanes 0-15 M=lane (K base 0), lanes 16-31 M=lane-16 (K+8);
// VGPR j<4 -> K=2j, j>=4 -> K=16+2(j-4).
__device__ inline v16h packA_f16(const _Float16* m, int r0, int lane) {
  H16 u;
  const int M = r0 + (lane & 15);
  const int koff = (lane >> 4) << 3;  // 0 or 8
#pragma unroll
  for (int j = 0; j < 8; ++j) {
    const int k0 = ((j & 4) << 2) + 2 * (j & 3) + koff;
    u.e[2 * j]     = m[M * 32 + k0];
    u.e[2 * j + 1] = m[M * 32 + k0 + 1];
  }
  return u.v;
}
__device__ inline v16h packA_f32(const float* m, int r0, int lane) {
  H16 u;
  const int M = r0 + (lane & 15);
  const int koff = (lane >> 4) << 3;
  const float2* row = (const float2*)(m + M * 32);  // 8B-aligned pairs
#pragma unroll
  for (int j = 0; j < 8; ++j) {
    const int k0 = ((j & 4) << 2) + 2 * (j & 3) + koff;  // always even
    float2 p = row[k0 >> 1];
    u.e[2 * j]     = (_Float16)p.x;
    u.e[2 * j + 1] = (_Float16)p.y;
  }
  return u.v;
}
// B (32x16 f16): lanes 0-15 N=lane (K base 0), lanes 16-31 N=lane-16 (K+8);
// VGPR j<4 -> K=2j, j>=4 -> K=16+2(j-4) (same K order as A).
__device__ inline v16h packB_f16(const _Float16* m, int c0, int lane) {
  H16 u;
  const int n = c0 + (lane & 15);
  const int koff = (lane >> 4) << 3;  // 0 or 8
#pragma unroll
  for (int j = 0; j < 8; ++j) {
    const int k0 = ((j & 4) << 2) + 2 * (j & 3) + koff;
    u.e[2 * j]     = m[k0 * 32 + n];
    u.e[2 * j + 1] = m[(k0 + 1) * 32 + n];
  }
  return u.v;
}
__device__ inline v16h packB_f32(const float* m, int c0, int lane) {
  H16 u;
  const int n = c0 + (lane & 15);
  const int koff = (lane >> 4) << 3;
#pragma unroll
  for (int j = 0; j < 8; ++j) {
    const int k0 = ((j & 4) << 2) + 2 * (j & 3) + koff;
    u.e[2 * j]     = (_Float16)m[k0 * 32 + n];
    u.e[2 * j + 1] = (_Float16)m[(k0 + 1) * 32 + n];
  }
  return u.v;
}
// D (16x16 f32): lanes 0-15 N=lane, M=vgpr; lanes 16-31 N=lane-16, M=8+vgpr.
__device__ inline void storeD(v8f acc, float* dst, int r0, int c0, int lane) {
  const int col = c0 + (lane & 15);
  const int rbase = r0 + ((lane >> 4) << 3);
#pragma unroll
  for (int r = 0; r < 8; ++r) dst[(rbase + r) * 32 + col] = acc[r];
}

// ---- constant DFT-table operand tiles, packed once per wave ---------------
struct ConstPack {
  v16h aC[2], aS[2], aSn[2];  // A-layout: rows 0-15 / 16-31 of cos, sin, -sin
  v16h bC[2], bS[2], bSn[2];  // B-layout: cols 0-15 / 16-31
};

// ---- 32x32 complex matmuls on WMMA ---------------------------------------
// Forward DFT32 = C - i*S = C + i*(-S): P=C, Q=Sn, Qn=S.
// Inverse DFT32 = C + i*S:              P=C, Q=S,  Qn=Sn.
// D = (P + i*Q) @ (U + i*V): Re = P@U + Qn@V ; Im = Q@U + P@V.
template <bool INV>
__device__ inline void mm_const_data(const ConstPack& T, const float* U,
                                     const float* V, float* Dre, float* Dim,
                                     int lane) {
#pragma unroll
  for (int ri = 0; ri < 2; ++ri) {
    const v16h aP  = T.aC[ri];
    const v16h aQ  = INV ? T.aS[ri] : T.aSn[ri];
    const v16h aQn = INV ? T.aSn[ri] : T.aS[ri];
#pragma unroll
    for (int ci = 0; ci < 2; ++ci) {
      const int c0 = ci << 4;
      const v16h bU = packB_f32(U, c0, lane);
      v8f accre = {}, accim = {};
      accre = WMMA(aP, bU, accre);
      accim = WMMA(aQ, bU, accim);
      if (V) {
        const v16h bV = packB_f32(V, c0, lane);
        accre = WMMA(aQn, bV, accre);
        accim = WMMA(aP, bV, accim);
      }
      storeD(accre, Dre, ri << 4, c0, lane);
      storeD(accim, Dim, ri << 4, c0, lane);
    }
  }
}
// D = (U + i*V) @ (P + i*Q): Re = U@P + V@Qn ; Im = U@Q + V@P.
// Dim may be null (inverse final stage only needs the real part).
template <bool INV>
__device__ inline void mm_data_const(const float* U, const float* V,
                                     const ConstPack& T, float* Dre,
                                     float* Dim, int lane) {
#pragma unroll
  for (int ri = 0; ri < 2; ++ri) {
    const int r0 = ri << 4;
    const v16h aU = packA_f32(U, r0, lane);
    const v16h aV = packA_f32(V, r0, lane);
#pragma unroll
    for (int ci = 0; ci < 2; ++ci) {
      const v16h bP  = T.bC[ci];
      const v16h bQ  = INV ? T.bS[ci] : T.bSn[ci];
      const v16h bQn = INV ? T.bSn[ci] : T.bS[ci];
      v8f accre = {};
      accre = WMMA(aU, bP, accre);
      accre = WMMA(aV, bQn, accre);
      if (Dim) {
        v8f accim = {};
        accim = WMMA(aU, bQ, accim);
        accim = WMMA(aV, bP, accim);
        storeD(accim, Dim, r0, ci << 4, lane);
      }
      storeD(accre, Dre, r0, ci << 4, lane);
    }
  }
}

// Step-2 twiddle: element [c][b] *= exp(sign*i*2pi*b*c/1024); lane = row c.
__device__ inline void twiddle(float* wr, float* wi, int lane, float sign) {
  const int c = lane;
#pragma unroll 4
  for (int bb = 0; bb < 32; ++bb) {
    const float ang = sign * 6.2831853072f * (float)(bb * c) * (1.0f / 1024.0f);
    const float cs = __cosf(ang), sn = __sinf(ang);
    const int idx = c * 32 + bb;
    const float re = wr[idx], im = wi[idx];
    wr[idx] = re * cs - im * sn;
    wi[idx] = re * sn + im * cs;
  }
}

// ---------------------------------------------------------------------------
// K0: zero output (harness poisons it; we accumulate with atomics).
__global__ void zero_kernel(float* __restrict__ p, int n) {
  for (int i = blockIdx.x * blockDim.x + threadIdx.x; i < n;
       i += gridDim.x * blockDim.x)
    p[i] = 0.0f;
}

// K1: per-batch exclusive scan of per-block phase increments.
// Block t of 512 samples of linearly-interpolated f0 sums in closed form:
// sum = 512*f0[t] + (f0[t+1]-f0[t])*255.5   (then /SR).
__global__ void prefix_kernel(const float* __restrict__ f0,
                              float* __restrict__ prefix) {
  __shared__ float bs[NFR];
  const int b = blockIdx.x;
  for (int t = threadIdx.x; t < NFR; t += blockDim.x) {
    const float a = f0[b * NFR + t];
    const float c = (t + 1 < NFR) ? f0[b * NFR + t + 1] : a;
    bs[t] = (512.0f * a + (c - a) * 255.5f) * (1.0f / FSR);
  }
  __syncthreads();
  if (threadIdx.x == 0) {
    float acc = 0.0f;
    for (int t = 0; t < NFR; ++t) {
      const float v = bs[t];
      bs[t] = acc;
      acc += v;
    }
  }
  __syncthreads();
  for (int t = threadIdx.x; t < NFR; t += blockDim.x)
    prefix[b * NFR + t] = bs[t];
}

// K2: combtooth = sinc(SR * wrap(cumsum(f0/SR)) / (f0+1e-3)).
// In-block inclusive cumsum of linearly-interpolated f0 is closed-form.
__global__ void comb_kernel(const float* __restrict__ f0,
                            const float* __restrict__ prefix,
                            float* __restrict__ comb) {
  const long total = (long)BATCH * NSMP;
  for (long i = (long)blockIdx.x * blockDim.x + threadIdx.x; i < total;
       i += (long)gridDim.x * blockDim.x) {
    const int b = (int)(i / NSMP);
    const int n = (int)(i % NSMP);
    const int t = n >> 9, j = n & 511;
    const float a = f0[b * NFR + t];
    const float c = (t + 1 < NFR) ? f0[b * NFR + t + 1] : a;
    const float frac = (float)j * (1.0f / 512.0f);
    const float f0n = a + (c - a) * frac;
    const float within =
        ((float)(j + 1) * a +
         (c - a) * (0.5f * (float)j * (float)(j + 1) * (1.0f / 512.0f))) *
        (1.0f / FSR);
    float x = prefix[b * NFR + t] + within;
    x -= rintf(x);
    const float z = FSR * x / (f0n + 1e-3f);
    const float pz = PI_F * z;
    comb[i] = (fabsf(pz) < 1e-5f) ? 1.0f : (__sinf(pz) / pz);
  }
}

// K3: per (batch,frame) wave: forward WMMA-FFTs of windowed comb & noise
// frames (TDM-loaded), spectral filtering, inverse WMMA-FFT, window,
// overlap-add. 2 waves/block, 8196 blocks == 8*2049 frames exactly (EXEC
// stays full at every WMMA).
__global__ __launch_bounds__(64) void synth_kernel(
    const float* __restrict__ comb, const float* __restrict__ noise,
    const float* __restrict__ hm, const float* __restrict__ hp,
    const float* __restrict__ nm, float* __restrict__ out) {
  __shared__ _Float16 Tc[1024], Tsin[1024], Tsn[1024];  // 32-pt DFT tables
  __shared__ float Win[NWIN];
  __shared__ float BR[2][1024], BI[2][1024];    // data / fft result
  __shared__ float WRr[2][1024], WIi[2][1024];  // four-step intermediate
  __shared__ float SPR[2][1024], SPI[2][1024];  // accumulated spectrum

  const int tid = threadIdx.x, lane = tid & 31, wv = tid >> 5;

  for (int i = tid; i < 1024; i += 64) {
    const int a = i >> 5, cc = i & 31;
    const float ang = 6.2831853072f * (float)(a * cc) * (1.0f / 32.0f);
    const float cs = __cosf(ang), sn = __sinf(ang);
    Tc[i] = (_Float16)cs;
    Tsin[i] = (_Float16)sn;
    Tsn[i] = (_Float16)(-sn);
    // sqrt(0.5-0.5cos(2pi i/1024)) == sin(pi i/1024)
    Win[i] = __sinf(PI_F * (float)i * (1.0f / 1024.0f));
  }
  __syncthreads();

  // Pack all constant WMMA operand tiles once; lives in VGPRs from here on.
  ConstPack T;
#pragma unroll
  for (int h = 0; h < 2; ++h) {
    T.aC[h]  = packA_f16(Tc,   h << 4, lane);
    T.aS[h]  = packA_f16(Tsin, h << 4, lane);
    T.aSn[h] = packA_f16(Tsn,  h << 4, lane);
    T.bC[h]  = packB_f16(Tc,   h << 4, lane);
    T.bS[h]  = packB_f16(Tsin, h << 4, lane);
    T.bSn[h] = packB_f16(Tsn,  h << 4, lane);
  }

  const int fid = blockIdx.x * 2 + wv;  // 0 .. 16391
  const int b = fid / (NFR + 1);
  const int t = fid % (NFR + 1);
  const int tt = (t < NFR) ? t : (NFR - 1);
  const size_t frow = (size_t)(b * NFR + tt) * NBINS;

  __builtin_prefetch(hm + frow, 0, 0);
  __builtin_prefetch(hp + frow, 0, 0);
  __builtin_prefetch(nm + frow, 0, 0);

  float* br = BR[wv];   float* bi = BI[wv];
  float* wr = WRr[wv];  float* wi = WIi[wv];
  float* spr = SPR[wv]; float* spi = SPI[wv];

  const float* combB = comb + (size_t)b * NSMP;
  const float* noiseB = noise + (size_t)b * NSMP;

  // ---- forward FFT of windowed comb frame (real input) ----
  load_frame(combB, br, t, Win, lane);  // M[a][b2] = frame[32a+b2] layout
  __syncthreads();
  mm_const_data<false>(T, br, nullptr, wr, wi, lane);  // F = cos - i sin
  __syncthreads();
  twiddle(wr, wi, lane, -1.0f);
  __syncthreads();
  mm_data_const<false>(wr, wi, T, br, bi, lane);
  __syncthreads();
  // bin k = c + 32d lives at [c*32+d]; apply src_filter, write spectrum +
  // Hermitian mirror (irfft equivalence).
  for (int k = lane; k <= BS; k += 32) {
    const float mg = __expf(hm[frow + k]);
    const float ph = PI_F * hp[frow + k];
    const float fr = mg * __cosf(ph), fi = mg * __sinf(ph);
    const int src = (k & 31) * 32 + (k >> 5);
    const float vr = br[src], vi = bi[src];
    const float orr = vr * fr - vi * fi;
    const float oii = vr * fi + vi * fr;
    spr[k] = orr;
    spi[k] = oii;
    if (k > 0 && k < BS) { spr[NWIN - k] = orr; spi[NWIN - k] = -oii; }
  }
  __syncthreads();

  // ---- forward FFT of windowed noise frame ----
  load_frame(noiseB, br, t, Win, lane);
  __syncthreads();
  mm_const_data<false>(T, br, nullptr, wr, wi, lane);
  __syncthreads();
  twiddle(wr, wi, lane, -1.0f);
  __syncthreads();
  mm_data_const<false>(wr, wi, T, br, bi, lane);
  __syncthreads();
  for (int k = lane; k <= BS; k += 32) {
    const float nf = __expf(nm[frow + k]) * (1.0f / 128.0f);  // real filter
    const int src = (k & 31) * 32 + (k >> 5);
    const float orr = br[src] * nf;
    const float oii = bi[src] * nf;
    spr[k] += orr;
    spi[k] += oii;
    if (k > 0 && k < BS) { spr[NWIN - k] += orr; spi[NWIN - k] += -oii; }
  }
  __syncthreads();

  // ---- inverse FFT (conjugate twiddles), real part only ----
  mm_const_data<true>(T, spr, spi, wr, wi, lane);  // Finv = cos + i sin
  __syncthreads();
  twiddle(wr, wi, lane, +1.0f);
  __syncthreads();
  mm_data_const<true>(wr, wi, T, br, nullptr, lane);
  __syncthreads();

  // ---- window + overlap-add: sample n = c + 32d at br[c*32+d] ----
  float* outB = out + (size_t)b * NSMP;
#pragma unroll 4
  for (int d = 0; d < 32; ++d) {
    const int n = lane + 32 * d;
    const int s = t * BS + n - BS;
    if (s >= 0 && s < NSMP) {
      const float v = br[lane * 32 + d] * (1.0f / 1024.0f) * Win[n];
      unsafeAtomicAdd(outB + s, v);  // global_atomic_add_f32
    }
  }
}

// ---------------------------------------------------------------------------
extern "C" void kernel_launch(void* const* d_in, const int* in_sizes, int n_in,
                              void* d_out, int out_size, void* d_ws,
                              size_t ws_size, hipStream_t stream) {
  (void)in_sizes; (void)n_in; (void)out_size; (void)ws_size;
  const float* f0    = (const float*)d_in[0];  // [8,2048]
  const float* hm    = (const float*)d_in[1];  // [8,2048,513]
  const float* hp    = (const float*)d_in[2];  // [8,2048,513]
  const float* nm    = (const float*)d_in[3];  // [8,2048,513]
  const float* noise = (const float*)d_in[4];  // [8,1048576]
  float* out = (float*)d_out;                  // [8,1048576]

  // workspace: prefix sums (8*2048 f32) then combtooth (8*1048576 f32)
  float* prefix = (float*)d_ws;
  float* comb =
      (float*)((char*)d_ws +
               (((size_t)BATCH * NFR * sizeof(float) + 255) & ~(size_t)255));

  zero_kernel<<<2048, 256, 0, stream>>>(out, BATCH * NSMP);
  prefix_kernel<<<BATCH, 256, 0, stream>>>(f0, prefix);
  comb_kernel<<<8192, 256, 0, stream>>>(f0, prefix, comb);
  synth_kernel<<<(BATCH * (NFR + 1)) / 2, 64, 0, stream>>>(comb, noise, hm, hp,
                                                           nm, out);
}
